// _QCALCGRTMLP_23880018165911
// MI455X (gfx1250) — matrix operand order repaired and checked by reading, not compiled, not run on hardware
//
#include <hip/hip_runtime.h>
#include <math.h>

// ---- problem constants ----
#define B_N   16384
#define F_N   256
#define NQ_N  2048
#define DT_N  256
#define TS_N  16
#define TK_N  8
#define GH_N  128
#define SH_N  128
#define SPD_N 272   // DT + TS
#define SPH_N 68
#define CD_N  278   // conf input dim
#define CH_N  69
#define RES_SCALE 0.1f

typedef _Float16 h16;
typedef __attribute__((ext_vector_type(16))) _Float16 v16h;
typedef __attribute__((ext_vector_type(8)))  float    v8f;

// -------------------- WMMA fragment helpers (gfx1250, wave32) --------------------
// A (MxK, f16): lane&15 = m, lane>>4 = half h.
//   VGPR v<4 : k = h*8 + 2v,2v+1       -> dword index h*4+v        (from &A[m][kbase])
//   VGPR v>=4: k = 16 + h*8 + 2(v-4).. -> dword index 8+h*4+(v-4)
__device__ __forceinline__ v16h fragA(const h16* p, int hh) {
  const uint32_t* d = (const uint32_t*)p;
  union { v16h v; uint32_t u[8]; } f;
#pragma unroll
  for (int i = 0; i < 4; ++i) f.u[i] = d[hh * 4 + i];
#pragma unroll
  for (int i = 0; i < 4; ++i) f.u[4 + i] = d[8 + hh * 4 + i];
  return f.v;
}
// B (KxN, f16) stored transposed wT[n][k]: lane&15 = n, same k order as A (k = h*8.. and 16+h*8..)
__device__ __forceinline__ v16h fragB(const h16* p, int hh) {
  const uint32_t* d = (const uint32_t*)p;
  union { v16h v; uint32_t u[8]; } f;
#pragma unroll
  for (int i = 0; i < 4; ++i) f.u[i] = d[hh * 4 + i];
#pragma unroll
  for (int i = 0; i < 4; ++i) f.u[4 + i] = d[8 + hh * 4 + i];
  return f.v;
}

__device__ __forceinline__ v8f wmma16(v16h a, v16h b, v8f c) {
  // D = A x B + C  :  v_wmma_f32_16x16x32_f16
  return __builtin_amdgcn_wmma_f32_16x16x32_f16(false, a, false, b, (short)0, c, false, false);
}

// One 16x16 output tile: A in LDS (stride As halves), wT in global (stride Ws halves)
__device__ __forceinline__ v8f gemm_tile(const h16* A, int As, const h16* WT, int Ws,
                                         int nt, int nkt, int lane) {
  const int ml = lane & 15;
  const int hh = (lane >> 4) & 1;
  const h16* arow = A + ml * As;
  const h16* wrow = WT + (size_t)(nt * 16 + ml) * Ws;
  v8f c = {};
  for (int kt = 0; kt < nkt; ++kt) {
    v16h a = fragA(arow + kt * 32, hh);
    v16h b = fragB(wrow + kt * 32, hh);
    c = wmma16(a, b, c);
  }
  return c;
}

// -------------------- kernel 1: weight transpose/convert to f16 --------------------
__global__ void prep_w(const float* __restrict__ src, h16* __restrict__ dst,
                       int K, int N, int Kpad, int Npad) {
  int i = blockIdx.x * 256 + threadIdx.x;
  if (i >= Kpad * Npad) return;
  int n = i / Kpad, k = i - n * Kpad;
  float v = (n < N && k < K) ? src[(size_t)k * N + n] : 0.f;
  dst[(size_t)n * Kpad + k] = (h16)v;
}

// -------------------- kernel 2: calibration views + stem MLP -> x_cal (f16) --------------------
__global__ __launch_bounds__(256) void calib_stem(
    const float* __restrict__ x_num, const float* __restrict__ center,
    const float* __restrict__ scale, const float* __restrict__ qs,
    const float* __restrict__ w1, const float* __restrict__ b1,
    const float* __restrict__ w2, const float* __restrict__ b2,
    const float* __restrict__ w3, const float* __restrict__ b3,
    h16* __restrict__ xcal) {
  __shared__ float sw1[48], sb1[16], sw2[256], sb2[16], sw3[16], sb3;
  int t = threadIdx.x;
  if (t < 48) sw1[t] = w1[t];
  if (t < 16) { sb1[t] = b1[t]; sb2[t] = b2[t]; sw3[t] = w3[t]; }
  sw2[t] = w2[t];
  if (t == 0) sb3 = b3[0];
  __syncthreads();

  size_t gid = (size_t)blockIdx.x * 256 + t;
  int f = (int)(gid & (F_N - 1));
  float x  = x_num[gid];
  float xr = (x - center[f]) / scale[f];
  // searchsorted (side='left') over column f of qs[NQ][F]
  int lo = 0, hi = NQ_N;
  while (lo < hi) {
    int mid = (lo + hi) >> 1;
    float qv = qs[(size_t)mid * F_N + f];
    if (qv < x) lo = mid + 1; else hi = mid;
  }
  float xq = fminf(fmaxf((float)lo * (1.f / (NQ_N - 1.f)), 0.f), 1.f);

  float h1[16];
#pragma unroll
  for (int j = 0; j < 16; ++j)
    h1[j] = fmaxf(x * sw1[j] + xr * sw1[16 + j] + xq * sw1[32 + j] + sb1[j], 0.f);
  float r = sb3;
#pragma unroll
  for (int j = 0; j < 16; ++j) {
    float h2 = sb2[j];
#pragma unroll
    for (int i = 0; i < 16; ++i) h2 += h1[i] * sw2[i * 16 + j];
    r += fmaxf(h2, 0.f) * sw3[j];
  }
  xcal[gid] = (h16)(x + RES_SCALE * tanhf(r));
}

// -------------------- kernel 3: fully fused per-16-row tile, one wave32 per block --------------------
struct WaveBuf {
  h16 sX[16 * 256];   // x_cal tile
  h16 sH[16 * 256];   // gate hidden, then h_base
  h16 sG[16 * 256];   // gate values
  h16 sA[16 * 288];   // staging A tile (x_gated / LN'd inputs), K padded to 288
  float sZ[16 * 16];  // z_top
  float gsum[16], gsum2[16], hsum[16], hsum2[16];
  float ybase[16], dsafe[16], dspec[16], dconf[16];
  float lnm[16], lnr[16];
  int gmaxI[16];
};

__global__ __launch_bounds__(32) void fused_tail(
    const h16* __restrict__ xcal,
    const h16* __restrict__ gw1T, const float* __restrict__ gate_b1,
    const h16* __restrict__ gw2T, const float* __restrict__ gate_b2,
    const h16* __restrict__ bbT,  const float* __restrict__ bb_b,
    const float* __restrict__ base_w, const float* __restrict__ base_b,
    const float* __restrict__ safe_lg, const float* __restrict__ safe_lb,
    const h16* __restrict__ sw1T, const float* __restrict__ safe_b1,
    const float* __restrict__ safe_w2, const float* __restrict__ safe_b2,
    const float* __restrict__ top_w1, const float* __restrict__ top_b1,
    const float* __restrict__ top_w2, const float* __restrict__ top_b2,
    const float* __restrict__ spec_lg, const float* __restrict__ spec_lb,
    const h16* __restrict__ spw1T, const float* __restrict__ spec_b1,
    const float* __restrict__ spec_w2, const float* __restrict__ spec_b2,
    const float* __restrict__ conf_lg, const float* __restrict__ conf_lb,
    const h16* __restrict__ cw1T, const float* __restrict__ conf_b1,
    const float* __restrict__ conf_w2, const float* __restrict__ conf_b2,
    float* __restrict__ out) {
  __shared__ WaveBuf wb;
  const int lane = threadIdx.x;
  const int n15 = lane & 15;
  const int hh = lane >> 4;
  const int rowBase = blockIdx.x * 16;

  // ---- load x_cal tile, zero accumulators ----
  {
    const uint32_t* src = (const uint32_t*)(xcal + (size_t)rowBase * 256);
    uint32_t* dst = (uint32_t*)wb.sX;
    for (int i = lane; i < 16 * 256 / 2; i += 32) dst[i] = src[i];
    if (lane < 16) {
      wb.gsum[lane] = 0.f; wb.gsum2[lane] = 0.f;
      wb.hsum[lane] = 0.f; wb.hsum2[lane] = 0.f;
      wb.ybase[lane] = 0.f; wb.dsafe[lane] = 0.f;
      wb.dspec[lane] = 0.f; wb.dconf[lane] = 0.f;
      wb.gmaxI[lane] = 0;  // g = sigmoid > 0, positive-float bits are monotone
    }
  }
  __syncthreads();

  // ---- gate layer 1: [16,256] @ [256,128], relu -> sH[:, :128] ----
  for (int nt = 0; nt < 8; ++nt) {
    v8f c = gemm_tile(wb.sX, 256, gw1T, 256, nt, 8, lane);
    int ng = nt * 16 + n15;
    float bn = gate_b1[ng];
#pragma unroll
    for (int r = 0; r < 8; ++r) {
      int mo = r + 8 * hh;
      wb.sH[mo * 256 + ng] = (h16)fmaxf(c[r] + bn, 0.f);
    }
  }
  __syncthreads();

  // ---- gate layer 2: [16,128] @ [128,256], sigmoid -> sG, + row stats ----
  for (int nt = 0; nt < 16; ++nt) {
    v8f c = gemm_tile(wb.sH, 256, gw2T, 128, nt, 4, lane);
    int ng = nt * 16 + n15;
    float bn = gate_b2[ng];
#pragma unroll
    for (int r = 0; r < 8; ++r) {
      int mo = r + 8 * hh;
      float g = 1.f / (1.f + expf(-(c[r] + bn)));
      wb.sG[mo * 256 + ng] = (h16)g;
      atomicAdd(&wb.gsum[mo], g);
      atomicAdd(&wb.gsum2[mo], g * g);
      atomicMax(&wb.gmaxI[mo], __float_as_int(g));
    }
  }
  __syncthreads();

  // ---- x_gated -> sA ----
  for (int i = lane; i < 16 * 256; i += 32) {
    int m = i >> 8, k = i & 255;
    wb.sA[m * 288 + k] = (h16)((float)wb.sX[i] * (float)wb.sG[i]);
  }
  __syncthreads();

  // ---- backbone: [16,256] @ [256,256], relu -> sH (h_base) + y_base + LN stats ----
  for (int nt = 0; nt < 16; ++nt) {
    v8f c = gemm_tile(wb.sA, 288, bbT, 256, nt, 8, lane);
    int ng = nt * 16 + n15;
    float bn = bb_b[ng], bw = base_w[ng];
#pragma unroll
    for (int r = 0; r < 8; ++r) {
      int mo = r + 8 * hh;
      float h = fmaxf(c[r] + bn, 0.f);
      wb.sH[mo * 256 + ng] = (h16)h;
      atomicAdd(&wb.hsum[mo], h);
      atomicAdd(&wb.hsum2[mo], h * h);
      atomicAdd(&wb.ybase[mo], h * bw);
    }
  }
  __syncthreads();

  // ---- safe branch: LN(h_base) -> sA ----
  if (lane < 16) {
    float mean = wb.hsum[lane] * (1.f / 256.f);
    float var  = wb.hsum2[lane] * (1.f / 256.f) - mean * mean;
    wb.lnm[lane] = mean;
    wb.lnr[lane] = rsqrtf(var + 1e-5f);
  }
  __syncthreads();
  for (int i = lane; i < 16 * 256; i += 32) {
    int m = i >> 8, k = i & 255;
    float hv = (float)wb.sH[i];
    wb.sA[m * 288 + k] = (h16)((hv - wb.lnm[m]) * wb.lnr[m] * safe_lg[k] + safe_lb[k]);
  }
  __syncthreads();
  // safe hidden GEMM + fused dot with safe_w2
  for (int nt = 0; nt < 8; ++nt) {
    v8f c = gemm_tile(wb.sA, 288, sw1T, 256, nt, 8, lane);
    int ng = nt * 16 + n15;
    float bn = safe_b1[ng], w2 = safe_w2[ng];
#pragma unroll
    for (int r = 0; r < 8; ++r)
      atomicAdd(&wb.dsafe[r + 8 * hh], fmaxf(c[r] + bn, 0.f) * w2);
  }
  __syncthreads();

  // ---- top-k gather + token MLP (one lane per row) ----
  if (lane < 16) {
    int m = lane;
    float vals[TK_N]; int idxs[TK_N];
    float prevV = 3.4e38f; int prevI = -1;
#pragma unroll
    for (int t = 0; t < TK_N; ++t) {
      float best = -3.4e38f; int bidx = 0;
      for (int f = 0; f < 256; ++f) {
        float v = (float)wb.sG[m * 256 + f];
        bool excl = (v > prevV) || (v == prevV && f <= prevI);
        if (!excl && v > best) { best = v; bidx = f; }
      }
      vals[t] = best; idxs[t] = bidx; prevV = best; prevI = bidx;
    }
    float s = 0.f;
#pragma unroll
    for (int t = 0; t < TK_N; ++t) s += vals[t];
    float invs = 1.f / (s + 1e-6f);
    float hacc[16];
#pragma unroll
    for (int j = 0; j < 16; ++j) hacc[j] = 0.f;
#pragma unroll
    for (int t = 0; t < TK_N; ++t) {
      float xw = (float)wb.sX[m * 256 + idxs[t]] * (vals[t] * invs);
#pragma unroll
      for (int j = 0; j < 16; ++j)
        hacc[j] += fmaxf(xw * top_w1[j] + top_b1[j], 0.f);
    }
#pragma unroll
    for (int o = 0; o < 16; ++o) {
      float z = 0.f;
#pragma unroll
      for (int j = 0; j < 16; ++j) z += hacc[j] * top_w2[j * 16 + o];
      wb.sZ[m * 16 + o] = z * (1.f / TK_N) + top_b2[o];
    }
  }
  __syncthreads();

  // ---- spec branch: LN(concat(h_base, z_top)) over 272 -> sA (K padded to 288) ----
  if (lane < 16) {
    int m = lane;
    float zs = 0.f, zq = 0.f;
#pragma unroll
    for (int o = 0; o < 16; ++o) { float z = wb.sZ[m * 16 + o]; zs += z; zq += z * z; }
    float mean = (wb.hsum[m] + zs) * (1.f / SPD_N);
    float var  = (wb.hsum2[m] + zq) * (1.f / SPD_N) - mean * mean;
    float rstd = rsqrtf(var + 1e-5f);
    wb.lnm[m] = mean; wb.lnr[m] = rstd;
    for (int k = 256; k < 288; ++k) {
      float v = 0.f;
      if (k < SPD_N) v = (wb.sZ[m * 16 + (k - 256)] - mean) * rstd * spec_lg[k] + spec_lb[k];
      wb.sA[m * 288 + k] = (h16)v;
    }
  }
  __syncthreads();
  for (int i = lane; i < 16 * 256; i += 32) {
    int m = i >> 8, k = i & 255;
    float hv = (float)wb.sH[i];
    wb.sA[m * 288 + k] = (h16)((hv - wb.lnm[m]) * wb.lnr[m] * spec_lg[k] + spec_lb[k]);
  }
  __syncthreads();
  for (int nt = 0; nt < 5; ++nt) {
    v8f c = gemm_tile(wb.sA, 288, spw1T, 288, nt, 9, lane);
    int ng = nt * 16 + n15;
    if (ng < SPH_N) {
      float bn = spec_b1[ng], w2 = spec_w2[ng];
#pragma unroll
      for (int r = 0; r < 8; ++r)
        atomicAdd(&wb.dspec[r + 8 * hh], fmaxf(c[r] + bn, 0.f) * w2);
    }
  }
  __syncthreads();

  // ---- conf branch: LN over 278-dim concat -> sA ----
  if (lane < 16) {
    int m = lane;
    float yb   = wb.ybase[m] + base_b[0];
    float ds   = wb.dsafe[m] + safe_b2[0];
    float dsp  = wb.dspec[m] + spec_b2[0];
    float gmean = wb.gsum[m] * (1.f / 256.f);
    float gmax  = __int_as_float(wb.gmaxI[m]);
    float gstd  = sqrtf(fmaxf(wb.gsum2[m] * (1.f / 256.f) - gmean * gmean, 0.f));
    float ads = fabsf(ds), adsp = fabsf(dsp);
    float zs = 0.f, zq = 0.f;
#pragma unroll
    for (int o = 0; o < 16; ++o) { float z = wb.sZ[m * 16 + o]; zs += z; zq += z * z; }
    float es = yb + gmean + gmax + gstd + ads + adsp + zs;
    float eq = yb * yb + gmean * gmean + gmax * gmax + gstd * gstd + ads * ads + adsp * adsp + zq;
    float mean = (wb.hsum[m] + es) * (1.f / CD_N);
    float var  = (wb.hsum2[m] + eq) * (1.f / CD_N) - mean * mean;
    float rstd = rsqrtf(var + 1e-5f);
    wb.lnm[m] = mean; wb.lnr[m] = rstd;
    for (int k = 256; k < 288; ++k) {
      float raw;
      if      (k == 256) raw = yb;
      else if (k == 257) raw = gmean;
      else if (k == 258) raw = gmax;
      else if (k == 259) raw = gstd;
      else if (k <  276) raw = wb.sZ[m * 16 + (k - 260)];
      else if (k == 276) raw = ads;
      else if (k == 277) raw = adsp;
      else               raw = 0.f;
      float v = (k < CD_N) ? ((raw - mean) * rstd * conf_lg[k] + conf_lb[k]) : 0.f;
      wb.sA[m * 288 + k] = (h16)v;
    }
  }
  __syncthreads();
  for (int i = lane; i < 16 * 256; i += 32) {
    int m = i >> 8, k = i & 255;
    float hv = (float)wb.sH[i];
    wb.sA[m * 288 + k] = (h16)((hv - wb.lnm[m]) * wb.lnr[m] * conf_lg[k] + conf_lb[k]);
  }
  __syncthreads();
  for (int nt = 0; nt < 5; ++nt) {
    v8f c = gemm_tile(wb.sA, 288, cw1T, 288, nt, 9, lane);
    int ng = nt * 16 + n15;
    if (ng < CH_N) {
      float bn = conf_b1[ng], w2 = conf_w2[ng];
#pragma unroll
      for (int r = 0; r < 8; ++r)
        atomicAdd(&wb.dconf[r + 8 * hh], fmaxf(c[r] + bn, 0.f) * w2);
    }
  }
  __syncthreads();

  // ---- final combine ----
  if (lane < 16) {
    int m = lane;
    float yb  = wb.ybase[m] + base_b[0];
    float ds  = wb.dsafe[m] + safe_b2[0];
    float dsp = wb.dspec[m] + spec_b2[0];
    float gamma = 1.f / (1.f + expf(-(wb.dconf[m] + conf_b2[0])));
    out[rowBase + m] = yb + ds + gamma * dsp;
  }
}

// -------------------- host launcher --------------------
extern "C" void kernel_launch(void* const* d_in, const int* in_sizes, int n_in,
                              void* d_out, int out_size, void* d_ws, size_t ws_size,
                              hipStream_t stream) {
  const float* x_num   = (const float*)d_in[0];
  const float* rcent   = (const float*)d_in[1];
  const float* rscale  = (const float*)d_in[2];
  const float* qs      = (const float*)d_in[3];
  const float* stem_w1 = (const float*)d_in[4];
  const float* stem_b1 = (const float*)d_in[5];
  const float* stem_w2 = (const float*)d_in[6];
  const float* stem_b2 = (const float*)d_in[7];
  const float* stem_w3 = (const float*)d_in[8];
  const float* stem_b3 = (const float*)d_in[9];
  const float* gate_w1 = (const float*)d_in[10];
  const float* gate_b1 = (const float*)d_in[11];
  const float* gate_w2 = (const float*)d_in[12];
  const float* gate_b2 = (const float*)d_in[13];
  const float* bb_w    = (const float*)d_in[14];
  const float* bb_b    = (const float*)d_in[15];
  const float* base_w  = (const float*)d_in[16];
  const float* base_b  = (const float*)d_in[17];
  const float* safe_lg = (const float*)d_in[18];
  const float* safe_lb = (const float*)d_in[19];
  const float* safe_w1 = (const float*)d_in[20];
  const float* safe_b1 = (const float*)d_in[21];
  const float* safe_w2 = (const float*)d_in[22];
  const float* safe_b2 = (const float*)d_in[23];
  const float* top_w1  = (const float*)d_in[24];
  const float* top_b1  = (const float*)d_in[25];
  const float* top_w2  = (const float*)d_in[26];
  const float* top_b2  = (const float*)d_in[27];
  const float* spec_lg = (const float*)d_in[28];
  const float* spec_lb = (const float*)d_in[29];
  const float* spec_w1 = (const float*)d_in[30];
  const float* spec_b1 = (const float*)d_in[31];
  const float* spec_w2 = (const float*)d_in[32];
  const float* spec_b2 = (const float*)d_in[33];
  const float* conf_lg = (const float*)d_in[34];
  const float* conf_lb = (const float*)d_in[35];
  const float* conf_w1 = (const float*)d_in[36];
  const float* conf_b1 = (const float*)d_in[37];
  const float* conf_w2 = (const float*)d_in[38];
  const float* conf_b2 = (const float*)d_in[39];
  float* out = (float*)d_out;

  // workspace carve-up (256B aligned)
  size_t off = 0;
  auto take = [&](size_t bytes) {
    size_t o = off;
    off += (bytes + 255) & ~(size_t)255;
    return o;
  };
  char* ws = (char*)d_ws;
  h16* xcal  = (h16*)(ws + take((size_t)B_N * F_N * 2));
  h16* gw1T  = (h16*)(ws + take((size_t)GH_N * 256 * 2));   // [128][256]
  h16* gw2T  = (h16*)(ws + take((size_t)256 * 128 * 2));    // [256][128]
  h16* bbT   = (h16*)(ws + take((size_t)256 * 256 * 2));    // [256][256]
  h16* sw1T  = (h16*)(ws + take((size_t)128 * 256 * 2));    // [128][256]
  h16* spw1T = (h16*)(ws + take((size_t)80 * 288 * 2));     // [80][288]
  h16* cw1T  = (h16*)(ws + take((size_t)80 * 288 * 2));     // [80][288]

  auto prep = [&](const float* src, h16* dst, int K, int N, int Kpad, int Npad) {
    int total = Kpad * Npad;
    prep_w<<<(total + 255) / 256, 256, 0, stream>>>(src, dst, K, N, Kpad, Npad);
  };
  prep(gate_w1, gw1T, 256, 128, 256, 128);
  prep(gate_w2, gw2T, 128, 256, 128, 256);
  prep(bb_w,   bbT,   256, 256, 256, 256);
  prep(safe_w1, sw1T, 256, 128, 256, 128);
  prep(spec_w1, spw1T, SPD_N, SPH_N, 288, 80);
  prep(conf_w1, cw1T,  CD_N,  CH_N,  288, 80);

  calib_stem<<<(B_N * F_N) / 256, 256, 0, stream>>>(
      x_num, rcent, rscale, qs,
      stem_w1, stem_b1, stem_w2, stem_b2, stem_w3, stem_b3, xcal);

  fused_tail<<<B_N / 16, 32, 0, stream>>>(
      xcal,
      gw1T, gate_b1, gw2T, gate_b2,
      bbT, bb_b, base_w, base_b,
      safe_lg, safe_lb, sw1T, safe_b1, safe_w2, safe_b2,
      top_w1, top_b1, top_w2, top_b2,
      spec_lg, spec_lb, spw1T, spec_b1, spec_w2, spec_b2,
      conf_lg, conf_lb, cw1T, conf_b1, conf_w2, conf_b2,
      out);
}
